// GnBlock_23493471109967
// MI455X (gfx1250) — matrix operand order repaired and checked by reading, not compiled, not run on hardware
//
#include <hip/hip_runtime.h>

typedef __attribute__((ext_vector_type(16))) _Float16 v16h;
typedef __attribute__((ext_vector_type(8)))  _Float16 v8h;
typedef __attribute__((ext_vector_type(8)))  float    v8f;

#define H 128
#define WAVES 4
#define HSTRIDE 136   // f16 elems per staged row: 272B, multiple of 16B -> aligned b128 LDS ops

// ---------------------------------------------------------------------------
// Pre-pack a (K x 128) f32 weight matrix into WMMA B-fragment order (f16).
// Fragment (kstep, ntile) = 512 f16, laid out [lane][j] so the main kernel
// loads 32 contiguous bytes per lane:
//   value = W[kstep*32 + ((j&8)<<1) + ((lane&16)>>1) + (j&7)][ntile*16 + (lane&15)]
// This matches the A-fragment k order: lanes 0-15 hold K=0..7,16..23,
// lanes 16-31 hold K=8..15,24..31, column = lane%16.
// ---------------------------------------------------------------------------
__global__ void pack_weights_kernel(const float* __restrict__ W,
                                    _Float16* __restrict__ out, int ksteps) {
  int idx = blockIdx.x * blockDim.x + threadIdx.x;
  int total = ksteps * 8 * 512;
  if (idx >= total) return;
  int frag = idx >> 9;
  int rem  = idx & 511;
  int lane = rem >> 4;
  int j    = rem & 15;
  int kstep = frag >> 3;
  int nt    = frag & 7;
  int k = kstep * 32 + ((j & 8) << 1) + ((lane & 16) >> 1) + (j & 7);
  int n = nt * 16 + (lane & 15);
  out[idx] = (_Float16)W[k * H + n];
}

// ---------------------------------------------------------------------------
// Fused  gather -> [3H->H] -> ReLU -> [H->H] -> LayerNorm -> residual store
//        (+ scatter-add of the normalized edge latent into agg, EDGE only)
// One wave computes one 16-row tile. wave32, EXEC all-ones in WMMA region.
// ---------------------------------------------------------------------------
template<bool EDGE>
__global__ __launch_bounds__(WAVES * 32)
void fused_mlp_ln_kernel(
    const float* __restrict__ c0, const float* __restrict__ c1,
    const float* __restrict__ c2,
    const int* __restrict__ sidx, const int* __restrict__ didx,
    const _Float16* __restrict__ w1p, const _Float16* __restrict__ w2p,
    const float* __restrict__ b1, const float* __restrict__ b2,
    const float* __restrict__ g,  const float* __restrict__ beta,
    const float* __restrict__ resid, float* __restrict__ out,
    float* __restrict__ agg, int ntiles, long nrows)
{
  __shared__ __align__(16) _Float16 hbuf[WAVES * 16 * HSTRIDE];

  const int wave = threadIdx.x >> 5;
  const int lane = threadIdx.x & 31;
  const int tile = blockIdx.x * WAVES + wave;
  if (tile >= ntiles) return;            // wave-uniform exit
  const int lanelo = lane & 15;
  const int hi8    = (lane >> 4) * 8;    // 0 or 8: K-group offset / row offset

  long grow = (long)tile * 16 + lanelo;  // A-matrix row owned by this lane
  if (grow >= nrows) grow = nrows - 1;

  // Row pointers for the three 128-wide chunks of the concatenated input.
  const float* rp0; const float* rp1; const float* rp2;
  if constexpr (EDGE) {
    rp0 = c0 + (long)sidx[grow] * H;     // x[sender]
    rp1 = c1 + (long)didx[grow] * H;     // x[receiver]
    rp2 = c2 + grow * H;                 // edge_attr
  } else {
    rp0 = c0 + grow * H;                 // x
    rp1 = c1 + grow * H;                 // agg_m
    rp2 = c2 + grow * H;                 // agg_w
  }

  // Per-lane column-parameters (col = nt*16 + lanelo), broadcast via cache.
  float b1v[8], b2v[8], gv[8], bev[8];
  #pragma unroll
  for (int nt = 0; nt < 8; ++nt) {
    int cn = nt * 16 + lanelo;
    b1v[nt] = b1[cn]; b2v[nt] = b2[cn]; gv[nt] = g[cn]; bev[nt] = beta[cn];
  }

  // ------------------- layer 1: (16x384) @ (384x128) + b1 -----------------
  v8f acc[8];
  #pragma unroll
  for (int nt = 0; nt < 8; ++nt)
    #pragma unroll
    for (int i = 0; i < 8; ++i) acc[nt][i] = b1v[nt];

  const float* rps[3] = { rp0, rp1, rp2 };
  #pragma unroll
  for (int c = 0; c < 3; ++c) {
    const float* p = rps[c];
    #pragma unroll
    for (int ks = 0; ks < 4; ++ks) {
      // A fragment 16x32 f16: lane holds K = {base..base+7, base+16..base+23},
      // base = ks*32 + hi8  (ISA 16-bit A layout).
      const float* q = p + ks * 32 + hi8;
      float4 f0 = *(const float4*)(q);
      float4 f1 = *(const float4*)(q + 4);
      float4 f2 = *(const float4*)(q + 16);
      float4 f3 = *(const float4*)(q + 20);
      v16h a;
      a[0]=(_Float16)f0.x; a[1]=(_Float16)f0.y; a[2]=(_Float16)f0.z; a[3]=(_Float16)f0.w;
      a[4]=(_Float16)f1.x; a[5]=(_Float16)f1.y; a[6]=(_Float16)f1.z; a[7]=(_Float16)f1.w;
      a[8]=(_Float16)f2.x; a[9]=(_Float16)f2.y; a[10]=(_Float16)f2.z; a[11]=(_Float16)f2.w;
      a[12]=(_Float16)f3.x; a[13]=(_Float16)f3.y; a[14]=(_Float16)f3.z; a[15]=(_Float16)f3.w;

      const _Float16* wb = w1p + (size_t)((c * 4 + ks) * 8) * 512 + lane * 16;
      #pragma unroll
      for (int nt = 0; nt < 8; ++nt) {
        v16h b = *(const v16h*)(wb + nt * 512);
        acc[nt] = __builtin_amdgcn_wmma_f32_16x16x32_f16(
            false, a, false, b, (short)0, acc[nt], false, false);
      }
    }
  }

  // -------- ReLU + stage hidden tile to LDS (C-layout -> row-major f16) ----
  _Float16* hb = hbuf + wave * 16 * HSTRIDE;
  #pragma unroll
  for (int nt = 0; nt < 8; ++nt)
    #pragma unroll
    for (int r = 0; r < 8; ++r) {
      float v = fmaxf(acc[nt][r], 0.0f);
      hb[(r + hi8) * HSTRIDE + nt * 16 + lanelo] = (_Float16)v;
    }
  // Cross-lane (intra-wave) transpose: make LDS stores visible before reload.
  asm volatile("s_wait_dscnt 0" ::: "memory");

  // ------------------- layer 2: (16x128) @ (128x128) + b2 -----------------
  v8f acc2[8];
  #pragma unroll
  for (int nt = 0; nt < 8; ++nt)
    #pragma unroll
    for (int i = 0; i < 8; ++i) acc2[nt][i] = b2v[nt];

  #pragma unroll
  for (int kk = 0; kk < 4; ++kk) {
    const _Float16* q = hb + lanelo * HSTRIDE + kk * 32 + hi8;
    v8h lo = *(const v8h*)(q);
    v8h hv = *(const v8h*)(q + 16);
    v16h a;
    #pragma unroll
    for (int j = 0; j < 8; ++j) { a[j] = lo[j]; a[8 + j] = hv[j]; }

    const _Float16* wb = w2p + (size_t)(kk * 8) * 512 + lane * 16;
    #pragma unroll
    for (int nt = 0; nt < 8; ++nt) {
      v16h b = *(const v16h*)(wb + nt * 512);
      acc2[nt] = __builtin_amdgcn_wmma_f32_16x16x32_f16(
          false, a, false, b, (short)0, acc2[nt], false, false);
    }
  }

  // ------------- LayerNorm row stats (shuffle over 16-lane halves) ---------
  // C/D layout: VGPR r, lanes 0-15 -> row r, lanes 16-31 -> row r+8.
  float mu[8], rs[8];
  #pragma unroll
  for (int r = 0; r < 8; ++r) {
    float s = 0.f, sq = 0.f;
    #pragma unroll
    for (int nt = 0; nt < 8; ++nt) { float v = acc2[nt][r]; s += v; sq += v * v; }
    #pragma unroll
    for (int m = 1; m < 16; m <<= 1) {
      s  += __shfl_xor(s,  m, 32);
      sq += __shfl_xor(sq, m, 32);
    }
    float mean = s * (1.0f / 128.0f);
    float var  = sq * (1.0f / 128.0f) - mean * mean;
    mu[r] = mean;
    rs[r] = rsqrtf(var + 1e-5f);
  }

  // ------------- normalize, residual store, fused scatter-add --------------
  #pragma unroll
  for (int r = 0; r < 8; ++r) {
    long orow = (long)tile * 16 + r + hi8;
    if (orow >= nrows) continue;
    const float* rr = resid + orow * H;
    float* oo = out + orow * H;
    float* ar = nullptr;
    if constexpr (EDGE) ar = agg + (long)didx[orow] * H;
    #pragma unroll
    for (int nt = 0; nt < 8; ++nt) {
      int cn = nt * 16 + lanelo;
      float v = (acc2[nt][r] - mu[r]) * rs[r] * gv[nt] + bev[nt];
      oo[cn] = rr[cn] + v;                   // residual output
      if constexpr (EDGE) atomicAdd(ar + cn, v);  // L2-resident segment-sum
    }
  }
}

// ---------------------------------------------------------------------------
// Host launcher
// inputs: 0 x, 1 edge_attr, 2 edge_world_attr,
//         3..8 emb_{W1,b1,W2,b2,g,beta}, 9..14 ewb_*, 15..20 nb_*,
//         21 edge_index(2xEM), 22 edge_world_index(2xEW)
// output: concat(x+xn, edge_attr+em, edge_world_attr+ew)
// ---------------------------------------------------------------------------
extern "C" void kernel_launch(void* const* d_in, const int* in_sizes, int n_in,
                              void* d_out, int out_size, void* d_ws, size_t ws_size,
                              hipStream_t stream) {
  (void)n_in; (void)out_size; (void)ws_size;

  const float* x  = (const float*)d_in[0];
  const float* ea = (const float*)d_in[1];
  const float* ew = (const float*)d_in[2];
  const float* P[3][6];
  for (int b = 0; b < 3; ++b)
    for (int j = 0; j < 6; ++j)
      P[b][j] = (const float*)d_in[3 + b * 6 + j];   // W1,b1,W2,b2,g,beta
  const int* eidx = (const int*)d_in[21];
  const int* widx = (const int*)d_in[22];

  const long N  = in_sizes[0] / H;
  const long EM = in_sizes[1] / H;
  const long EW = in_sizes[2] / H;
  const int* es   = eidx;      const int* er   = eidx + EM;
  const int* wsrc = widx;      const int* wrcv = widx + EW;

  // Workspace carve: packed weights (3 x 128KB) + two N x 128 f32 agg buffers.
  char* wsp = (char*)d_ws;
  size_t off = 0;
  auto carve = [&](size_t bytes) -> char* {
    char* p = wsp + off;
    off = (off + bytes + 255) & ~(size_t)255;
    return p;
  };
  _Float16* w1p[3]; _Float16* w2p[3];
  for (int b = 0; b < 3; ++b) {
    w1p[b] = (_Float16*)carve((size_t)12 * 8 * 512 * sizeof(_Float16));
    w2p[b] = (_Float16*)carve((size_t)4  * 8 * 512 * sizeof(_Float16));
  }
  float* aggm = (float*)carve((size_t)N * H * sizeof(float));
  float* aggw = (float*)carve((size_t)N * H * sizeof(float));

  hipMemsetAsync(aggm, 0, (size_t)N * H * sizeof(float), stream);
  hipMemsetAsync(aggw, 0, (size_t)N * H * sizeof(float), stream);

  for (int b = 0; b < 3; ++b) {
    pack_weights_kernel<<<(12 * 8 * 512 + 255) / 256, 256, 0, stream>>>(P[b][0], w1p[b], 12);
    pack_weights_kernel<<<( 4 * 8 * 512 + 255) / 256, 256, 0, stream>>>(P[b][2], w2p[b], 4);
  }

  float* out_x = (float*)d_out;
  float* out_e = out_x + (size_t)N * H;
  float* out_w = out_e + (size_t)EM * H;

  {   // EdgeMeshBlock (scatter into aggm)
    int tiles = (int)((EM + 15) / 16);
    dim3 gridd((tiles + WAVES - 1) / WAVES);
    fused_mlp_ln_kernel<true><<<gridd, WAVES * 32, 0, stream>>>(
        x, x, ea, es, er, w1p[0], w2p[0],
        P[0][1], P[0][3], P[0][4], P[0][5],
        ea, out_e, aggm, tiles, EM);
  }
  {   // EdgeWorldBlock (scatter into aggw)
    int tiles = (int)((EW + 15) / 16);
    dim3 gridd((tiles + WAVES - 1) / WAVES);
    fused_mlp_ln_kernel<true><<<gridd, WAVES * 32, 0, stream>>>(
        x, x, ew, wsrc, wrcv, w1p[1], w2p[1],
        P[1][1], P[1][3], P[1][4], P[1][5],
        ew, out_w, aggw, tiles, EW);
  }
  {   // NodeBlock (consumes agg buffers; stream order guarantees visibility)
    int tiles = (int)((N + 15) / 16);
    dim3 gridd((tiles + WAVES - 1) / WAVES);
    fused_mlp_ln_kernel<false><<<gridd, WAVES * 32, 0, stream>>>(
        x, aggm, aggw, nullptr, nullptr, w1p[2], w2p[2],
        P[2][1], P[2][3], P[2][4], P[2][5],
        x, out_x, nullptr, tiles, N);
  }
}
